// gnn_atomic_lg_final_10179072491637
// MI455X (gfx1250) — matrix operand order repaired and checked by reading, not compiled, not run on hardware
//
#include <hip/hip_runtime.h>

// ---------------------------------------------------------------------------
// Problem constants (B=1, N=4096, M=16384, J=3, F0=F1=32, C=2)
// ---------------------------------------------------------------------------
#define NN      4096
#define MM      16384
#define KW      (NN * 3)      // 12288 floats per W row  (k = m*3 + j)
#define KP      (MM * 2)      // 32768 floats per P row  (k = m*2 + jp)
#define W_ELEMS ((size_t)NN * KW)   // 50,331,648 floats of W in d_out

typedef __attribute__((ext_vector_type(16))) _Float16 v16h;
typedef __attribute__((ext_vector_type(8)))  float    v8f;
typedef __attribute__((ext_vector_type(4)))  float    f4;

// ---------------------------------------------------------------------------
// Phase A: WMMA projection.  out2[(m*JDIV + j)*2 + c] = sum_f X[m,f]*Wp[c, j*32+f]
// One wave (32 threads) per 16-row tile.  Uses v_wmma_f32_16x16x32_f16:
//   A = 16x32 f16 (16 rows of X), B = 32x16 f16 (columns = (c,j) pairs),
//   D = 16x16 f32, only first NCOLS columns are meaningful.
// Fragment layouts: A/B in contiguous K order (lane half h holds K = 16h..16h+15); f32 C/D in hardware layout.
// NCOLS/JDIV/WSTRIDE are compile-time so no integer division is emitted.
// ---------------------------------------------------------------------------
template <int NCOLS, int JDIV, int WSTRIDE>
__global__ __launch_bounds__(32)
void wmma_proj_kernel(const float* __restrict__ X,     // [rows, 32]
                      const float* __restrict__ Wp,    // [C, JDIV*32]
                      float* __restrict__ out2)        // [rows*JDIV, 2] interleaved c
{
    const int lane = threadIdx.x;        // 0..31
    const int half = lane >> 4;          // K half select
    const int l16  = lane & 15;

    // ---- A fragment: row = tile*16 + l16 ----
    const int row = blockIdx.x * 16 + l16;
    const float* xr = X + (size_t)row * 32;
    const int fb0 = half ? 16 : 0;       // halfs 0..7  hold K = half*16 + 0..7
    const int fb1 = half ? 24 : 8;       // halfs 8..15 hold K = half*16 + 8..15
    v16h a;
    #pragma unroll
    for (int i = 0; i < 8; ++i) {
        a[i]     = (_Float16)xr[fb0 + i];
        a[8 + i] = (_Float16)xr[fb1 + i];
    }

    // ---- B fragment: column N = l16, K range = half*16 .. half*16+15 ----
    const int c = l16 / JDIV;            // compile-time divisor -> shifts/muls
    const int j = l16 % JDIV;
    v16h b;
    if (l16 < NCOLS) {
        const float* wr = Wp + c * WSTRIDE + j * 32 + half * 16;
        #pragma unroll
        for (int i = 0; i < 16; ++i) b[i] = (_Float16)wr[i];
    } else {
        #pragma unroll
        for (int i = 0; i < 16; ++i) b[i] = (_Float16)0.0f;
    }

    v8f cacc = {};
    v8f d = __builtin_amdgcn_wmma_f32_16x16x32_f16(
        /*neg_a=*/false, a, /*neg_b=*/false, b,
        /*c_mod=*/(short)0, cacc, /*reuse_a=*/false, /*reuse_b=*/false);

    // ---- D: VGPR r -> (M = r or 8+r, N = l16); store interleaved (c0,c1) ----
    if (l16 < NCOLS) {
        const int mbase = blockIdx.x * 16 + (half ? 8 : 0);
        float* dst = out2 + ((size_t)mbase * JDIV + j) * 2 + c;
        #pragma unroll
        for (int r = 0; r < 8; ++r) {
            dst[(size_t)r * (JDIV * 2)] = d[r];   // constant stride per row
        }
    }
}

// ---------------------------------------------------------------------------
// Phase B: bandwidth-bound streaming reduction + fused W copy.
// One block per row n.  Reads W row (48 KB) and P row (128 KB) once with
// non-temporal float4 loads (stream-once data; keep the 352 KB coefficient
// tables resident in the 192 MB L2), writes the W row to d_out with NT stores,
// and accumulates the two dot products (c=0, c=1).
// ---------------------------------------------------------------------------
__global__ __launch_bounds__(256)
void stream_reduce_kernel(const float* __restrict__ W,
                          const float* __restrict__ P,
                          const float* __restrict__ uu2,   // [KW,2]
                          const float* __restrict__ vv2,   // [KP,2]
                          const float* __restrict__ bx,
                          const float* __restrict__ by,
                          float* __restrict__ Wout,
                          float* __restrict__ xy)
{
    const int n   = blockIdx.x;
    const int tid = threadIdx.x;

    const f4* __restrict__ Wrow = (const f4*)(W + (size_t)n * KW);   // 3072 f4
    const f4* __restrict__ Prow = (const f4*)(P + (size_t)n * KP);   // 8192 f4
    f4* __restrict__ WoutRow    = (f4*)(Wout + (size_t)n * KW);
    const f4* __restrict__ U    = (const f4*)uu2;                    // 2 f4 per f4 of W
    const f4* __restrict__ V    = (const f4*)vv2;

    float acc0 = 0.0f, acc1 = 0.0f;

    #pragma unroll 4
    for (int i = tid; i < KW / 4; i += 256) {
        const f4 w   = __builtin_nontemporal_load(&Wrow[i]);
        const f4 c01 = U[2 * i + 0];   // (c0,c1) for k=4i and 4i+1
        const f4 c23 = U[2 * i + 1];   // (c0,c1) for k=4i+2 and 4i+3
        acc0 += w.x * c01.x + w.y * c01.z + w.z * c23.x + w.w * c23.z;
        acc1 += w.x * c01.y + w.y * c01.w + w.z * c23.y + w.w * c23.w;
        __builtin_nontemporal_store(w, &WoutRow[i]);   // fused copy of W to d_out
    }

    #pragma unroll 4
    for (int i = tid; i < KP / 4; i += 256) {
        const f4 p   = __builtin_nontemporal_load(&Prow[i]);
        const f4 c01 = V[2 * i + 0];
        const f4 c23 = V[2 * i + 1];
        acc0 += p.x * c01.x + p.y * c01.z + p.z * c23.x + p.w * c23.z;
        acc1 += p.x * c01.y + p.y * c01.w + p.z * c23.y + p.w * c23.w;
    }

    __shared__ float s0[256];
    __shared__ float s1[256];
    s0[tid] = acc0;
    s1[tid] = acc1;
    __syncthreads();
    #pragma unroll
    for (int s = 128; s > 0; s >>= 1) {
        if (tid < s) { s0[tid] += s0[tid + s]; s1[tid] += s1[tid + s]; }
        __syncthreads();
    }
    if (tid == 0) {
        xy[(size_t)n * 2 + 0] = s0[0] + bx[0] + by[0];
        xy[(size_t)n * 2 + 1] = s1[0] + bx[1] + by[1];
    }
}

// ---------------------------------------------------------------------------
// Launch.  Inputs (setup_inputs order):
//   0:W [1,4096,4096,3]  1:x [1,4096,32]  2:W_lg (unused)  3:y [1,16384,32]
//   4:P [1,4096,16384,2] 5:Wx2x [2,96]    6:bx2x [2]       7:Wy2x [2,64]  8:by2x [2]
// d_out = W (50,331,648 f32) ++ xy2x (8192 f32)
// ---------------------------------------------------------------------------
extern "C" void kernel_launch(void* const* d_in, const int* in_sizes, int n_in,
                              void* d_out, int out_size, void* d_ws, size_t ws_size,
                              hipStream_t stream) {
    (void)in_sizes; (void)n_in; (void)out_size; (void)ws_size;

    const float* W    = (const float*)d_in[0];
    const float* x    = (const float*)d_in[1];
    const float* y    = (const float*)d_in[3];
    const float* P    = (const float*)d_in[4];
    const float* Wx2x = (const float*)d_in[5];
    const float* bx2x = (const float*)d_in[6];
    const float* Wy2x = (const float*)d_in[7];
    const float* by2x = (const float*)d_in[8];

    float* out  = (float*)d_out;
    float* Wout = out;
    float* xy   = out + W_ELEMS;

    // Workspace: uu2 = KW*2 floats (96 KB), vv2 = KP*2 floats (256 KB)
    float* uu2 = (float*)d_ws;
    float* vv2 = (float*)((char*)d_ws + (size_t)KW * 2 * sizeof(float));

    // Phase A: WMMA projections (tiny: 256 + 1024 waves)
    wmma_proj_kernel<6, 3, 96><<<NN / 16, 32, 0, stream>>>(x, Wx2x, uu2);
    wmma_proj_kernel<4, 2, 64><<<MM / 16, 32, 0, stream>>>(y, Wy2x, vv2);

    // Phase B: stream W + P once at HBM rate, fused W copy, emit xy2x.
    stream_reduce_kernel<<<NN, 256, 0, stream>>>(W, P, uu2, vv2, bx2x, by2x, Wout, xy);
}
